// Net_13494787244871
// MI455X (gfx1250) — matrix operand order repaired and checked by reading, not compiled, not run on hardware
//
#include <hip/hip_runtime.h>
#include <hip/hip_bf16.h>
#include <math.h>

typedef __attribute__((ext_vector_type(16))) _Float16 v16h;
typedef __attribute__((ext_vector_type(8)))  _Float16 v8h;
typedef __attribute__((ext_vector_type(8)))  float    v8f;

#define DEV __device__ __forceinline__

// ---------------------------------------------------------------------------
// WMMA fragment helpers (CDNA5 16x16x32 f16, wave32; A/B k order is contiguous, C/D per ISA 7.12.2)
// A 16x32: lane&15 = row M; halfs0-15 = K kb..kb+15
//          where kb = k0 + (lane>=16 ? 16 : 0)
// B 32x16: lane&15 = col N; lanes 0-15 hold K k0..k0+15, lanes 16-31 K k0+16..31
// C/D 16x16: lane&15 = col N; vgpr i = row m0 + i (+8 for lanes 16-31)
// ---------------------------------------------------------------------------

DEV v16h frag_zero() { v16h z = {}; return z; }

DEV v16h load_a16(const _Float16* __restrict__ A, int ldk, int m0, int k0, int lane) {
    int row = m0 + (lane & 15);
    int kb  = k0 + ((lane >> 4) << 4);
    const _Float16* p = A + (size_t)row * ldk + kb;
    v8h lo = *(const v8h*)(p);
    v8h hi = *(const v8h*)(p + 8);
    v16h r;
#pragma unroll
    for (int i = 0; i < 8; ++i) { r[i] = lo[i]; r[i + 8] = hi[i]; }
    return r;
}

DEV v16h load_b16(const _Float16* __restrict__ Wm, int ldk, int n0, int k0, int lane) {
    int col = n0 + (lane & 15);
    int kk  = k0 + ((lane >> 4) << 4);
    return *(const v16h*)(Wm + (size_t)col * ldk + kk);
}

DEV v8f wmma16(v16h a, v16h b, v8f c) {
    return __builtin_amdgcn_wmma_f32_16x16x32_f16(false, a, false, b, (short)0, c,
                                                  false, false);
}

// ---------------------------------------------------------------------------
// Weight preparation
// ---------------------------------------------------------------------------

__global__ void k_f32_to_f16(const float* __restrict__ s, _Float16* __restrict__ d, size_t n) {
    size_t i = (size_t)blockIdx.x * 256 + threadIdx.x;
    if (i < n) d[i] = (_Float16)s[i];
}

// OIHW [O][C][3][3] fp32 -> [tap][O][C] f16  (tap = ky*3+kx)
__global__ void k_pack_conv(const float* __restrict__ w, _Float16* __restrict__ wp,
                            int O, int C) {
    size_t idx = (size_t)blockIdx.x * 256 + threadIdx.x;
    size_t total = (size_t)O * C * 9;
    if (idx >= total) return;
    int t = (int)(idx % 9);
    size_t oc = idx / 9;
    wp[(size_t)t * O * C + oc] = (_Float16)w[idx];
}

// ---------------------------------------------------------------------------
// conv1: K=27 too small for WMMA -> direct, NCHW f32 in, NHWC f16 out, ReLU
// ---------------------------------------------------------------------------

__global__ void k_conv1(const float* __restrict__ img, const float* __restrict__ w,
                        const float* __restrict__ b, _Float16* __restrict__ Y) {
    size_t idx = (size_t)blockIdx.x * 256 + threadIdx.x;
    if (idx >= (size_t)16 * 224 * 224 * 64) return;
    int oc = (int)(idx & 63);
    size_t p = idx >> 6;
    int x = (int)(p % 224);
    size_t q = p / 224;
    int y = (int)(q % 224);
    int bi = (int)(q / 224);
    float acc = b[oc];
    for (int ci = 0; ci < 3; ++ci) {
        const float* ip = img + ((size_t)(bi * 3 + ci)) * 224 * 224;
        const float* wp = w + ((size_t)oc * 3 + ci) * 9;
#pragma unroll
        for (int ky = 0; ky < 3; ++ky) {
            int yy = y + ky - 1;
            if (yy < 0 || yy >= 224) continue;
#pragma unroll
            for (int kx = 0; kx < 3; ++kx) {
                int xx = x + kx - 1;
                if (xx < 0 || xx >= 224) continue;
                acc += ip[(size_t)yy * 224 + xx] * wp[ky * 3 + kx];
            }
        }
    }
    acc = acc > 0.f ? acc : 0.f;
    Y[idx] = (_Float16)acc;
}

// ---------------------------------------------------------------------------
// 3x3 conv as 9 shifted implicit GEMMs via WMMA. X: NHWC f16, Wp: [9][Cout][Cin]
// One wave per 16(pixels) x 64(channels) tile: one A fragment feeds 4 WMMAs.
// Requires Cout % 64 == 0. Fused bias + ReLU.
// ---------------------------------------------------------------------------

__global__ void __launch_bounds__(32)
k_conv3x3(const _Float16* __restrict__ X, const _Float16* __restrict__ Wp,
          const float* __restrict__ bias, _Float16* __restrict__ Y,
          int H, int Wd, int Cin, int Cout) {
    int lane = threadIdx.x;
    int m0 = blockIdx.x << 4;
    int n0 = blockIdx.y << 6;              // 64 output channels per wave
    int mrow = m0 + (lane & 15);
    int xx0 = mrow % Wd;
    int t1 = mrow / Wd;
    int yy0 = t1 % H;
    int bi = t1 / H;
    v8f acc[4] = {};
#pragma unroll
    for (int t = 0; t < 9; ++t) {
        int dy = t / 3 - 1, dx = t % 3 - 1;
        int yy = yy0 + dy, xx = xx0 + dx;
        bool valid = (yy >= 0) && (yy < H) && (xx >= 0) && (xx < Wd);
        const _Float16* basep = X + ((size_t)((bi * H + yy) * Wd + xx)) * Cin;
        const _Float16* Wt = Wp + (size_t)t * Cout * Cin;
        for (int c0 = 0; c0 < Cin; c0 += 32) {
            v16h a;
            if (valid) {
                int kb = c0 + ((lane >> 4) << 4);
                v8h lo = *(const v8h*)(basep + kb);
                v8h hi = *(const v8h*)(basep + kb + 8);
#pragma unroll
                for (int i = 0; i < 8; ++i) { a[i] = lo[i]; a[i + 8] = hi[i]; }
            } else {
                a = frag_zero();
            }
#pragma unroll
            for (int j = 0; j < 4; ++j)
                acc[j] = wmma16(a, load_b16(Wt, Cin, n0 + (j << 4), c0, lane), acc[j]);
        }
    }
    int rb = m0 + ((lane >> 4) << 3);
#pragma unroll
    for (int j = 0; j < 4; ++j) {
        int col = n0 + (j << 4) + (lane & 15);
        float bv = bias[col];
#pragma unroll
        for (int i = 0; i < 8; ++i) {
            float v = acc[j][i] + bv;
            v = v > 0.f ? v : 0.f;
            Y[(size_t)(rb + i) * Cout + col] = (_Float16)v;
        }
    }
}

// 2x2 max pool, NHWC f16
__global__ void k_maxpool(const _Float16* __restrict__ X, _Float16* __restrict__ Y,
                          int Ho, int Wo, int C) {
    size_t idx = (size_t)blockIdx.x * 256 + threadIdx.x;
    size_t total = (size_t)16 * Ho * Wo * C;
    if (idx >= total) return;
    int c = (int)(idx % C);
    size_t p = idx / C;
    int xo = (int)(p % Wo); p /= Wo;
    int yo = (int)(p % Ho);
    int b = (int)(p / Ho);
    int Wi = 2 * Wo, Hi = 2 * Ho;
    const _Float16* rp = X + (((size_t)(b * Hi + 2 * yo)) * Wi + 2 * xo) * C + c;
    float a0 = (float)rp[0];
    float a1 = (float)rp[C];
    float a2 = (float)rp[(size_t)Wi * C];
    float a3 = (float)rp[(size_t)Wi * C + C];
    Y[idx] = (_Float16)fmaxf(fmaxf(a0, a1), fmaxf(a2, a3));
}

// global average pool over 28x28, NHWC [16,28,28,512] -> [16,512] f16
__global__ void k_gap(const _Float16* __restrict__ X, _Float16* __restrict__ feat) {
    int idx = blockIdx.x * 256 + threadIdx.x;
    if (idx >= 16 * 512) return;
    int c = idx % 512, b = idx / 512;
    const _Float16* p = X + (size_t)b * 28 * 28 * 512 + c;
    float s = 0.f;
    for (int i = 0; i < 784; ++i) s += (float)p[(size_t)i * 512];
    feat[idx] = (_Float16)(s * (1.0f / 784.0f));
}

// ---------------------------------------------------------------------------
// WMMA GEMM, 4 N-tiles per wave: C[M,N] = A[M,K] * W[N,K]^T + bias  (f32 out)
// blockIdx.x = M tile (16 rows), blockIdx.y = N block (64 cols). N%? handled
// by host-side tail launch of the 1-tile variant. Weight rows prefetched.
// ---------------------------------------------------------------------------

__global__ void __launch_bounds__(32)
k_gemm_bias4(const _Float16* __restrict__ A, const _Float16* __restrict__ Wm,
             const float* __restrict__ bias, float* __restrict__ C, int N, int K) {
    int lane = threadIdx.x;
    int m0 = blockIdx.x << 4;
    int n0 = blockIdx.y << 6;
    v8f acc[4] = {};
    for (int k0 = 0; k0 < K; k0 += 32) {
        v16h a = load_a16(A, K, m0, k0, lane);
        if (k0 + 128 < K) {
            // global_prefetch_b8 of the weight stream 4 K-steps ahead
            int col = n0 + (lane & 15);
            __builtin_prefetch(Wm + (size_t)col * K + k0 + 128, 0, 1);
        }
#pragma unroll
        for (int j = 0; j < 4; ++j)
            acc[j] = wmma16(a, load_b16(Wm, K, n0 + (j << 4), k0, lane), acc[j]);
    }
    int rb = m0 + ((lane >> 4) << 3);
#pragma unroll
    for (int j = 0; j < 4; ++j) {
        int col = n0 + (j << 4) + (lane & 15);
        float bv = bias ? bias[col] : 0.0f;
#pragma unroll
        for (int i = 0; i < 8; ++i) C[(size_t)(rb + i) * N + col] = acc[j][i] + bv;
    }
}

// single-tile variant with column offset, for N-tail handling
__global__ void __launch_bounds__(32)
k_gemm_bias1(const _Float16* __restrict__ A, const _Float16* __restrict__ Wm,
             const float* __restrict__ bias, float* __restrict__ C, int N, int K,
             int n_base) {
    int lane = threadIdx.x;
    int m0 = blockIdx.x << 4;
    int n0 = n_base + (blockIdx.y << 4);
    v8f acc = {};
    for (int k0 = 0; k0 < K; k0 += 32)
        acc = wmma16(load_a16(A, K, m0, k0, lane), load_b16(Wm, K, n0, k0, lane), acc);
    int col = n0 + (lane & 15);
    float bv = bias ? bias[col] : 0.0f;
    int rb = m0 + ((lane >> 4) << 3);
#pragma unroll
    for (int i = 0; i < 8; ++i) C[(size_t)(rb + i) * N + col] = acc[i] + bv;
}

// gates[16,2560] = xin[16,1280]*Wih^T + h16[16,640]*Whh^T + b_ih + b_hh
// 4 N-tiles per wave (2560/64 = 40 blocks)
__global__ void __launch_bounds__(32)
k_gates4(const _Float16* __restrict__ xin, const _Float16* __restrict__ wih,
         const _Float16* __restrict__ h16, const _Float16* __restrict__ whh,
         const float* __restrict__ bih, const float* __restrict__ bhh,
         float* __restrict__ gates) {
    int lane = threadIdx.x;
    int n0 = blockIdx.x << 6;
    v8f acc[4] = {};
    for (int k0 = 0; k0 < 1280; k0 += 32) {
        v16h a = load_a16(xin, 1280, 0, k0, lane);
#pragma unroll
        for (int j = 0; j < 4; ++j)
            acc[j] = wmma16(a, load_b16(wih, 1280, n0 + (j << 4), k0, lane), acc[j]);
    }
    for (int k0 = 0; k0 < 640; k0 += 32) {
        v16h a = load_a16(h16, 640, 0, k0, lane);
#pragma unroll
        for (int j = 0; j < 4; ++j)
            acc[j] = wmma16(a, load_b16(whh, 640, n0 + (j << 4), k0, lane), acc[j]);
    }
    int rb = (lane >> 4) << 3;
#pragma unroll
    for (int j = 0; j < 4; ++j) {
        int col = n0 + (j << 4) + (lane & 15);
        float bv = bih[col] + bhh[col];
#pragma unroll
        for (int i = 0; i < 8; ++i) gates[(size_t)(rb + i) * 2560 + col] = acc[j][i] + bv;
    }
}

// ---------------------------------------------------------------------------
// LSTM glue
// ---------------------------------------------------------------------------

// e = emb[captions[:,t]]; fused = [e | memory]
__global__ void k_prep_fused(const float* __restrict__ emb, const int* __restrict__ captions,
                             int t, const _Float16* __restrict__ mem16,
                             _Float16* __restrict__ e16, _Float16* __restrict__ fused) {
    int idx = blockIdx.x * 256 + threadIdx.x;
    if (idx >= 16 * 640) return;
    int bb = idx / 640, j = idx % 640;
    int tok = captions[bb * 32 + t];
    _Float16 ev = (_Float16)emb[(size_t)tok * 640 + j];
    e16[idx] = ev;
    fused[(size_t)bb * 1280 + j] = ev;
    fused[(size_t)bb * 1280 + 640 + j] = mem16[idx];
}

// softmax(logits) per row, context = attn*memory, xin = [e | context]
__global__ void k_softmax_ctx(const float* __restrict__ logits, const float* __restrict__ mem,
                              const _Float16* __restrict__ e16, _Float16* __restrict__ xin) {
    int bb = blockIdx.x;
    int tid = threadIdx.x;
    __shared__ float red[256];
    const float* lrow = logits + (size_t)bb * 640;
    float m = -1e30f;
    for (int j = tid; j < 640; j += 256) m = fmaxf(m, lrow[j]);
    red[tid] = m; __syncthreads();
    for (int s = 128; s > 0; s >>= 1) {
        if (tid < s) red[tid] = fmaxf(red[tid], red[tid + s]);
        __syncthreads();
    }
    m = red[0]; __syncthreads();
    float sum = 0.f;
    for (int j = tid; j < 640; j += 256) sum += expf(lrow[j] - m);
    red[tid] = sum; __syncthreads();
    for (int s = 128; s > 0; s >>= 1) {
        if (tid < s) red[tid] += red[tid + s];
        __syncthreads();
    }
    float inv = 1.0f / red[0];
    for (int j = tid; j < 640; j += 256) {
        float attn = expf(lrow[j] - m) * inv;
        float ctx = attn * mem[(size_t)bb * 640 + j];
        xin[(size_t)bb * 1280 + j] = e16[(size_t)bb * 640 + j];
        xin[(size_t)bb * 1280 + 640 + j] = (_Float16)ctx;
    }
}

DEV float sigmoidf(float x) { return 1.0f / (1.0f + expf(-x)); }

__global__ void k_lstm_update(const float* __restrict__ gates, float* __restrict__ c,
                              _Float16* __restrict__ h16, _Float16* __restrict__ outs16,
                              int t) {
    int idx = blockIdx.x * 256 + threadIdx.x;
    if (idx >= 16 * 640) return;
    int bb = idx / 640, j = idx % 640;
    const float* g = gates + (size_t)bb * 2560;
    float ig = sigmoidf(g[j]);
    float fg = sigmoidf(g[640 + j]);
    float gg = tanhf(g[1280 + j]);
    float og = sigmoidf(g[1920 + j]);
    float cc = fg * c[idx] + ig * gg;
    c[idx] = cc;
    float hh = og * tanhf(cc);
    h16[idx] = (_Float16)hh;
    outs16[((size_t)bb * 32 + t) * 640 + j] = (_Float16)hh;   // [B,T,H] order
}

// ---------------------------------------------------------------------------
// Host orchestration
// ---------------------------------------------------------------------------

extern "C" void kernel_launch(void* const* d_in, const int* in_sizes, int n_in,
                              void* d_out, int out_size, void* d_ws, size_t ws_size,
                              hipStream_t stream) {
    (void)in_sizes; (void)n_in; (void)out_size; (void)ws_size;
    const float* images  = (const float*)d_in[0];
    const int*   captions= (const int*)d_in[1];
    const float* cw1 = (const float*)d_in[2];
    const float* cb1 = (const float*)d_in[3];
    const float* cw2 = (const float*)d_in[4];
    const float* cb2 = (const float*)d_in[5];
    const float* cw3 = (const float*)d_in[6];
    const float* cb3 = (const float*)d_in[7];
    const float* cw4 = (const float*)d_in[8];
    const float* cb4 = (const float*)d_in[9];
    const float* enc_w = (const float*)d_in[10];
    const float* enc_b = (const float*)d_in[11];
    const float* emb   = (const float*)d_in[12];
    const float* attn_w= (const float*)d_in[13];
    const float* attn_b= (const float*)d_in[14];
    const float* w_ih  = (const float*)d_in[15];
    const float* w_hh  = (const float*)d_in[16];
    const float* b_ih  = (const float*)d_in[17];
    const float* b_hh  = (const float*)d_in[18];
    const float* fc_w  = (const float*)d_in[19];
    const float* fc_b  = (const float*)d_in[20];
    float* out = (float*)d_out;

    char* base = (char*)d_ws;
    size_t off = 0;
    auto alloc = [&](size_t bytes) -> void* {
        void* p = base + off;
        off += (bytes + 255) & ~(size_t)255;
        return p;
    };

    _Float16* actA   = (_Float16*)alloc((size_t)16 * 224 * 224 * 64 * 2); // ping
    _Float16* actB   = (_Float16*)alloc((size_t)16 * 112 * 112 * 64 * 2); // pong
    _Float16* wp2    = (_Float16*)alloc((size_t)9 * 128 * 64 * 2);
    _Float16* wp3    = (_Float16*)alloc((size_t)9 * 256 * 128 * 2);
    _Float16* wp4    = (_Float16*)alloc((size_t)9 * 512 * 256 * 2);
    _Float16* encw16 = (_Float16*)alloc((size_t)640 * 512 * 2);
    _Float16* attnw16= (_Float16*)alloc((size_t)640 * 1280 * 2);
    _Float16* wih16  = (_Float16*)alloc((size_t)2560 * 1280 * 2);
    _Float16* whh16  = (_Float16*)alloc((size_t)2560 * 640 * 2);
    _Float16* fcw16  = (_Float16*)alloc((size_t)10000 * 640 * 2);
    _Float16* feat16 = (_Float16*)alloc((size_t)16 * 512 * 2);
    float*    memf   = (float*)   alloc((size_t)16 * 640 * 4);
    _Float16* mem16  = (_Float16*)alloc((size_t)16 * 640 * 2);
    _Float16* e16    = (_Float16*)alloc((size_t)16 * 640 * 2);
    _Float16* fused16= (_Float16*)alloc((size_t)16 * 1280 * 2);
    float*    logits = (float*)   alloc((size_t)16 * 640 * 4);
    _Float16* xin16  = (_Float16*)alloc((size_t)16 * 1280 * 2);
    float*    gatesb = (float*)   alloc((size_t)16 * 2560 * 4);
    float*    cf     = (float*)   alloc((size_t)16 * 640 * 4);
    _Float16* h16    = (_Float16*)alloc((size_t)16 * 640 * 2);
    _Float16* outs16 = (_Float16*)alloc((size_t)512 * 640 * 2);

    dim3 b256(256);
    auto g1 = [](size_t n) { return dim3((unsigned)((n + 255) / 256)); };

    // ---- weight prep (f32 -> f16, conv taps repacked [9][O][C]) ----
    k_pack_conv<<<g1((size_t)128 * 64 * 9),  b256, 0, stream>>>(cw2, wp2, 128, 64);
    k_pack_conv<<<g1((size_t)256 * 128 * 9), b256, 0, stream>>>(cw3, wp3, 256, 128);
    k_pack_conv<<<g1((size_t)512 * 256 * 9), b256, 0, stream>>>(cw4, wp4, 512, 256);
    k_f32_to_f16<<<g1((size_t)640 * 512),    b256, 0, stream>>>(enc_w,  encw16, (size_t)640 * 512);
    k_f32_to_f16<<<g1((size_t)640 * 1280),   b256, 0, stream>>>(attn_w, attnw16,(size_t)640 * 1280);
    k_f32_to_f16<<<g1((size_t)2560 * 1280),  b256, 0, stream>>>(w_ih,   wih16,  (size_t)2560 * 1280);
    k_f32_to_f16<<<g1((size_t)2560 * 640),   b256, 0, stream>>>(w_hh,   whh16,  (size_t)2560 * 640);
    k_f32_to_f16<<<g1((size_t)10000 * 640),  b256, 0, stream>>>(fc_w,   fcw16,  (size_t)10000 * 640);

    // ---- CNN encoder (NHWC f16 activations) ----
    k_conv1<<<g1((size_t)16 * 224 * 224 * 64), b256, 0, stream>>>(images, cw1, cb1, actA);
    k_maxpool<<<g1((size_t)16 * 112 * 112 * 64), b256, 0, stream>>>(actA, actB, 112, 112, 64);
    k_conv3x3<<<dim3(12544, 2), dim3(32), 0, stream>>>(actB, wp2, cb2, actA, 112, 112, 64, 128);
    k_maxpool<<<g1((size_t)16 * 56 * 56 * 128), b256, 0, stream>>>(actA, actB, 56, 56, 128);
    k_conv3x3<<<dim3(3136, 4),  dim3(32), 0, stream>>>(actB, wp3, cb3, actA, 56, 56, 128, 256);
    k_maxpool<<<g1((size_t)16 * 28 * 28 * 256), b256, 0, stream>>>(actA, actB, 28, 28, 256);
    k_conv3x3<<<dim3(784, 8),   dim3(32), 0, stream>>>(actB, wp4, cb4, actA, 28, 28, 256, 512);
    k_gap<<<g1((size_t)16 * 512), b256, 0, stream>>>(actA, feat16);

    // memory = feat @ enc_w^T + enc_b   (M=16, N=640, K=512)
    k_gemm_bias4<<<dim3(1, 10), dim3(32), 0, stream>>>(feat16, encw16, enc_b, memf, 640, 512);
    k_f32_to_f16<<<g1((size_t)16 * 640), b256, 0, stream>>>(memf, mem16, (size_t)16 * 640);

    hipMemsetAsync(cf,  0, (size_t)16 * 640 * 4, stream);
    hipMemsetAsync(h16, 0, (size_t)16 * 640 * 2, stream);

    // ---- LSTM decode, 32 sequential steps ----
    for (int t = 0; t < 32; ++t) {
        k_prep_fused<<<g1((size_t)16 * 640), b256, 0, stream>>>(emb, captions, t, mem16, e16, fused16);
        // logits = fused @ attn_w^T + attn_b  (M=16, N=640, K=1280)
        k_gemm_bias4<<<dim3(1, 10), dim3(32), 0, stream>>>(fused16, attnw16, attn_b, logits, 640, 1280);
        k_softmax_ctx<<<dim3(16), b256, 0, stream>>>(logits, memf, e16, xin16);
        // gates = xin@Wih^T + h@Whh^T + biases  (M=16, N=2560)
        k_gates4<<<dim3(40), dim3(32), 0, stream>>>(xin16, wih16, h16, whh16, b_ih, b_hh, gatesb);
        k_lstm_update<<<g1((size_t)16 * 640), b256, 0, stream>>>(gatesb, cf, h16, outs16, t);
    }

    // ---- output projection: [512,640] x [10000,640]^T + fc_b -> d_out ----
    // body: 156 blocks of 64 cols (9984), tail: one 16-col tile at 9984
    k_gemm_bias4<<<dim3(32, 156), dim3(32), 0, stream>>>(outs16, fcw16, fc_b, out, 10000, 640);
    k_gemm_bias1<<<dim3(32, 1),   dim3(32), 0, stream>>>(outs16, fcw16, fc_b, out, 10000, 640, 9984);
}
